// IntProject_86586540687648
// MI455X (gfx1250) — 8-bit matrix operand order repaired and checked by reading, compiled, not run on hardware
//
#include <hip/hip_runtime.h>

typedef __attribute__((ext_vector_type(8))) int v8i;
typedef __attribute__((ext_vector_type(4))) int v4i;
typedef __attribute__((ext_vector_type(2))) int v2i;

#define DK 4096
#define DN 4096
#define BM 128
#define BN 256
#define BK 64
#define NK (DK / BK)

#if __has_builtin(__builtin_amdgcn_global_load_async_to_lds_b128)
#define HAS_ASYNC 1
typedef int vs4i __attribute__((vector_size(16)));
typedef __attribute__((address_space(1))) vs4i gvs4i;
typedef __attribute__((address_space(3))) vs4i lvs4i;
__device__ __forceinline__ void async_b128(const signed char* g, const signed char* l) {
  __builtin_amdgcn_global_load_async_to_lds_b128(
      (gvs4i*)(unsigned long long)g,
      (lvs4i*)(unsigned)(unsigned long long)l,
      0, 0);
}
#if __has_builtin(__builtin_amdgcn_s_wait_asynccnt)
#define WAIT_ASYNC(n) __builtin_amdgcn_s_wait_asynccnt(n)
#else
#define WAIT_ASYNC(n) asm volatile("s_wait_asynccnt %0" ::"n"(n) : "memory")
#endif
#else
#define HAS_ASYNC 0
#endif

__global__ __launch_bounds__(256) void pack_x_kernel(const int* __restrict__ x,
                                                     int* __restrict__ xp, int n4) {
  int i = blockIdx.x * blockDim.x + threadIdx.x;
  if (i >= n4) return;
  v4i v = ((const v4i*)x)[i];
  int packed = (v.x & 0xff) | ((v.y & 0xff) << 8) | ((v.z & 0xff) << 16) | (v.w << 24);
  xp[i] = packed;
}

__global__ __launch_bounds__(256) void transpose_w_kernel(const signed char* __restrict__ W,
                                                          signed char* __restrict__ Wt) {
  __shared__ signed char t[64][80];
  const int bn = blockIdx.x * 64;
  const int bk = blockIdx.y * 64;
  const int tid = threadIdx.x;
  const int r  = tid >> 2;
  const int c  = (tid & 3) * 16;

  v4i v = *(const v4i*)(W + (size_t)(bk + r) * DN + bn + c);
  *(v4i*)(&t[r][c]) = v;
  __syncthreads();

  int dst[4];
  signed char* db = (signed char*)dst;
#pragma unroll
  for (int j = 0; j < 16; ++j) db[j] = t[c + j][r];
  *(v4i*)(Wt + (size_t)(bn + r) * DK + bk + c) = *(v4i*)dst;
}

__global__ __launch_bounds__(256) void gemm_iu8_kernel(const signed char* __restrict__ A,
                                                       const signed char* __restrict__ Bt,
                                                       const int* __restrict__ bias,
                                                       const int* __restrict__ shift_p,
                                                       int* __restrict__ out) {
  __shared__ signed char lA[2][BM * BK];
  __shared__ signed char lB[2][BN * BK];

  const int tid   = threadIdx.x;
  const int lane  = tid & 31;
  const int wave  = tid >> 5;
  const int wm    = wave >> 2;
  const int wn    = wave & 3;
  const int m0    = blockIdx.y * BM;
  const int n0    = blockIdx.x * BN;
  const int lhalf = lane >> 4;
  const int l15   = lane & 15;

  const int ra = tid >> 1;
  const int ca = (tid & 1) * 32;
  const signed char* gA = A  + (size_t)(m0 + ra)  * DK + ca;
  const signed char* gB = Bt + (size_t)(n0 + tid) * DK;

  v8i acc[4][4] = {};

  auto issue = [&](int ktile, int stage) {
    const int k = ktile * BK;
#if HAS_ASYNC
    async_b128(gA + k,      &lA[stage][ra * BK + ca]);
    async_b128(gA + k + 16, &lA[stage][ra * BK + ca + 16]);
    async_b128(gB + k,      &lB[stage][tid * BK]);
    async_b128(gB + k + 16, &lB[stage][tid * BK + 16]);
    async_b128(gB + k + 32, &lB[stage][tid * BK + 32]);
    async_b128(gB + k + 48, &lB[stage][tid * BK + 48]);
#else
    v4i a0 = *(const v4i*)(gA + k);
    v4i a1 = *(const v4i*)(gA + k + 16);
    v4i b0 = *(const v4i*)(gB + k);
    v4i b1 = *(const v4i*)(gB + k + 16);
    v4i b2 = *(const v4i*)(gB + k + 32);
    v4i b3 = *(const v4i*)(gB + k + 48);
    *(v4i*)&lA[stage][ra * BK + ca]      = a0;
    *(v4i*)&lA[stage][ra * BK + ca + 16] = a1;
    *(v4i*)&lB[stage][tid * BK]          = b0;
    *(v4i*)&lB[stage][tid * BK + 16]     = b1;
    *(v4i*)&lB[stage][tid * BK + 32]     = b2;
    *(v4i*)&lB[stage][tid * BK + 48]     = b3;
#endif
  };

  issue(0, 0);

  for (int i = 0; i < NK; ++i) {
    const int cur = i & 1;
    if (i + 1 < NK) {
      issue(i + 1, cur ^ 1);
#if HAS_ASYNC
      WAIT_ASYNC(6);
#endif
    } else {
#if HAS_ASYNC
      WAIT_ASYNC(0);
#endif
    }
    __syncthreads();

    v8i bfrag[4];
#pragma unroll
    for (int nt = 0; nt < 4; ++nt) {
      const signed char* p = &lB[cur][(wn * 64 + nt * 16 + l15) * BK + lhalf * 16];
      v4i lo = *(const v4i*)(p);
      v4i hi = *(const v4i*)(p + 32);
      v8i b;
      b[0] = lo.x; b[1] = lo.y; b[2] = lo.z; b[3] = lo.w;
      b[4] = hi.x; b[5] = hi.y; b[6] = hi.z; b[7] = hi.w;
      bfrag[nt] = b;
    }

#pragma unroll
    for (int mt = 0; mt < 4; ++mt) {
      const signed char* p = &lA[cur][(wm * 64 + mt * 16 + l15) * BK + lhalf * 16];
      v2i p0 = *(const v2i*)(p);
      v2i p1 = *(const v2i*)(p + 8);
      v2i p2 = *(const v2i*)(p + 32);
      v2i p3 = *(const v2i*)(p + 40);
      v8i a;
      a[0] = p0.x; a[1] = p0.y; a[2] = p1.x; a[3] = p1.y;
      a[4] = p2.x; a[5] = p2.y; a[6] = p3.x; a[7] = p3.y;
#pragma unroll
      for (int nt = 0; nt < 4; ++nt) {
        acc[mt][nt] = __builtin_amdgcn_wmma_i32_16x16x64_iu8(
            true, a, true, bfrag[nt], acc[mt][nt], false, false);
      }
    }
    __syncthreads();
  }

  const int shift = *shift_p;
#pragma unroll
  for (int mt = 0; mt < 4; ++mt) {
#pragma unroll
    for (int nt = 0; nt < 4; ++nt) {
      const int col = n0 + wn * 64 + nt * 16 + l15;
      const int bv  = bias[col];
      const int rowbase = m0 + wm * 64 + mt * 16 + lhalf * 8;
#pragma unroll
      for (int j = 0; j < 8; ++j) {
        int y = (acc[mt][nt][j] + bv) >> shift;
        out[(size_t)(rowbase + j) * DN + col] = y;
      }
    }
  }
}

extern "C" void kernel_launch(void* const* d_in, const int* in_sizes, int n_in,
                              void* d_out, int out_size, void* d_ws, size_t ws_size,
                              hipStream_t stream) {
  const int*         x     = (const int*)d_in[0];
  const signed char* w     = (const signed char*)d_in[1];
  const int*         bias  = (const int*)d_in[2];
  const int*         shift = (const int*)d_in[3];
  int*               out   = (int*)d_out;

  const size_t M = (size_t)in_sizes[0] / DK;

  signed char* xp = (signed char*)d_ws;
  signed char* wt = xp + M * DK;

  const int n4 = (int)(M * DK / 4);
  pack_x_kernel<<<(n4 + 255) / 256, 256, 0, stream>>>(x, (int*)xp, n4);
  transpose_w_kernel<<<dim3(DN / 64, DK / 64), 256, 0, stream>>>(w, wt);
  gemm_iu8_kernel<<<dim3(DN / BN, (unsigned)(M / BM)), 256, 0, stream>>>(
      xp, wt, bias, shift, out);
}
